// CharCountRNN_9466107920460
// MI455X (gfx1250) — matrix operand order repaired and checked by reading, not compiled, not run on hardware
//
#include <hip/hip_runtime.h>
#include <hip/hip_bf16.h>

// ---------------------------------------------------------------------------
// CDNA5 (gfx1250) bidirectional 2-layer GRU.
//   - feed-forward GEMMs + recurrent GEMMs on v_wmma_f32_16x16x32_f16
//   - batch-parallel scan: one workgroup owns 16 batch rows for all 256 steps
//   - W_hh lives in registers (12 B-frags/wave), h exchanged via double-
//     buffered LDS tile, xp prefetched ahead of the WMMA chain.
// ---------------------------------------------------------------------------

typedef __attribute__((ext_vector_type(16))) _Float16 v16h;
typedef __attribute__((ext_vector_type(8)))  _Float16 v8h;
typedef __attribute__((ext_vector_type(8)))  float    v8f;

#define TT     256
#define BB     1024
#define HH     128
#define G3     384          // 3*H
#define NTIL   24           // 384/16
#define IN0P   416          // 400 padded to multiple of 32
#define IN1    256
#define MROWS  (TT*BB)      // 262144
#define MTILES (MROWS/16)   // 16384
#define BTILES (BB/16)      // 64

__device__ __forceinline__ v8f wmma16(v16h a, v16h b, v8f c) {
  return __builtin_amdgcn_wmma_f32_16x16x32_f16(false, a, false, b, (short)0, c,
                                                false, false);
}

// A-fragment (16x32, MxK) from row-major f16, per CDNA5 ISA 16-bit A layout:
// lanes 0-15: row M=lane, K = 0..7 & 16..23 ; lanes 16-31: row M=lane-16,
// K = 8..15 & 24..31.
__device__ __forceinline__ v16h load_a_frag(const _Float16* a, int ldk) {
  int lane = threadIdx.x & 31;
  int m    = lane & 15;
  int sel  = lane >> 4;
  const _Float16* p = a + (size_t)m * ldk + sel * 8;
  v8h lo = *(const v8h*)(p);        // K = sel*8 .. sel*8+7
  v8h hi = *(const v8h*)(p + 16);   // K = 16+sel*8 ..
  v16h r;
#pragma unroll
  for (int i = 0; i < 8; ++i) { r[i] = lo[i]; r[i + 8] = hi[i]; }
  return r;
}

// B-fragment (32x16, KxN) from an N x K row-major weight matrix:
// lanes 0-15: col N=lane, K = 0..7 & 16..23 ; lanes 16-31: col N=lane-16,
// K = 8..15 & 24..31.
__device__ __forceinline__ v16h load_b_frag(const _Float16* w, int ldk) {
  int lane = threadIdx.x & 31;
  int n    = lane & 15;
  int sel  = lane >> 4;
  const _Float16* p = w + (size_t)n * ldk + sel * 8;
  v8h lo = *(const v8h*)(p);
  v8h hi = *(const v8h*)(p + 16);
  v16h r;
#pragma unroll
  for (int i = 0; i < 8; ++i) { r[i] = lo[i]; r[i + 8] = hi[i]; }
  return r;
}

__device__ __forceinline__ float fast_sigmoid(float x) {
  return 1.0f / (1.0f + __expf(-x));
}
__device__ __forceinline__ float fast_tanh(float x) {
  float a = fabsf(x);
  float e = __expf(-2.0f * a);
  float r = (1.0f - e) / (1.0f + e);
  return copysignf(r, x);
}

// --------------------------- weight fp32 -> f16 ----------------------------
__global__ void cvt_w_kernel(const float* __restrict__ src,
                             _Float16* __restrict__ dst,
                             int N, int Ksrc, int Kdst) {
  int idx = blockIdx.x * blockDim.x + threadIdx.x;
  if (idx >= N * Kdst) return;
  int n = idx / Kdst, k = idx - n * Kdst;
  float v = (k < Ksrc) ? src[(size_t)n * Ksrc + k] : 0.0f;
  dst[idx] = (_Float16)v;
}

// ------------------- feature build: argmax gather + concat -----------------
// one wave per (t,b) row; lane-parallel argmax (first-max tie rule), then
// cooperative f16 writes of the 416-wide padded feature row.
__global__ void build_x_kernel(const float* __restrict__ input,
                               const float* __restrict__ g_style,
                               const float* __restrict__ char_style,
                               _Float16* __restrict__ xh) {
  int wave = blockIdx.x * (blockDim.x >> 5) + (threadIdx.x >> 5);
  int lane = threadIdx.x & 31;
  size_t row = (size_t)wave;          // row = t*B + b
  int b = (int)(row & (BB - 1));
  const float* in_row = input + row * 80;

  float bv = -3.402823e38f; int bi = 0;
#pragma unroll
  for (int s = 0; s < 3; ++s) {
    int c = lane + 32 * s;
    if (c < 80) {
      float v = in_row[c];
      if (v > bv) { bv = v; bi = c; }
    }
  }
#pragma unroll
  for (int m = 16; m > 0; m >>= 1) {
    float ov = __shfl_xor(bv, m);
    int   oi = __shfl_xor(bi, m);
    if (ov > bv || (ov == bv && oi < bi)) { bv = ov; bi = oi; }
  }

  const float* gs = g_style + (size_t)b * 256;
  const float* cs = char_style + ((size_t)b * 80 + bi) * 64;
  _Float16* xrow = xh + row * IN0P;
#pragma unroll
  for (int s = 0; s < 13; ++s) {
    int c = lane + 32 * s;
    float v;
    if      (c < 80)  v = in_row[c];
    else if (c < 336) v = gs[c - 80];
    else if (c < 400) v = cs[c - 336];
    else              v = 0.0f;
    xrow[c] = (_Float16)v;
  }
}

// --------------------- input-projection GEMM (WMMA) ------------------------
// xp = A(Mx K) @ W^T(K x 384) + b_ih ; A row-major f16, W packed N x K f16.
// 8 waves/block, each wave owns one 16-row M-tile across all 24 N-tiles
// (24 v8f accumulators = 192 VGPRs; CDNA5 budget is 1024). Output is stored
// in C-fragment tile order so the scan kernel reads it back coalesced.
__global__ void __launch_bounds__(256)
gemm_xp_kernel(const _Float16* __restrict__ A,
               const _Float16* __restrict__ Wt,   // 384 x K, row-major f16
               const float* __restrict__ bias,    // 384 fp32
               _Float16* __restrict__ xp,         // MTILES*24*256 f16 frags
               int K, int mtiles_per_launch) {
  int wave  = threadIdx.x >> 5;
  int lane  = threadIdx.x & 31;
  int mtile = blockIdx.x * 8 + wave;
  (void)mtiles_per_launch;

  const _Float16* abase = A + (size_t)mtile * 16 * K;
  v8f acc[NTIL] = {};

  int ktiles = K >> 5;
  for (int kt = 0; kt < ktiles; ++kt) {
    v16h a = load_a_frag(abase + kt * 32, K);
#pragma unroll
    for (int nt = 0; nt < NTIL; ++nt) {
      v16h bfr = load_b_frag(Wt + (size_t)(nt * 16) * K + kt * 32, K);
      acc[nt] = wmma16(a, bfr, acc[nt]);
    }
  }

  int col = lane & 15;
#pragma unroll
  for (int nt = 0; nt < NTIL; ++nt) {
    float bsum = bias[nt * 16 + col];
    v8h o;
#pragma unroll
    for (int i = 0; i < 8; ++i) o[i] = (_Float16)(acc[nt][i] + bsum);
    *(v8h*)(xp + ((size_t)mtile * NTIL + nt) * 256 + lane * 8) = o;
  }
}

// ------------------------------ GRU scan -----------------------------------
// grid = (64 batch-tiles, 2 directions); block = 256 (8 waves).
// Wave w owns N-tiles {w, 8+w, 16+w} -> gates r,z,n for h-columns 16w..16w+15,
// so each wave produces one 16x16 tile of h_new locally every step.
__global__ void __launch_bounds__(256)
gru_scan_kernel(const _Float16* __restrict__ xp_f,
                const _Float16* __restrict__ xp_r,
                const _Float16* __restrict__ whh_f,  // 384 x 128 f16
                const _Float16* __restrict__ whh_r,
                const float* __restrict__ bhh_f,
                const float* __restrict__ bhh_r,
                _Float16* __restrict__ y) {          // MROWS x 256, row-major
  int btile = blockIdx.x;
  int dir   = blockIdx.y;
  int wave  = threadIdx.x >> 5;
  int lane  = threadIdx.x & 31;
  int col   = lane & 15;
  int mrow  = (lane >> 4) * 8;

  const _Float16* xp  = dir ? xp_r  : xp_f;
  const _Float16* whh = dir ? whh_r : whh_f;
  const float*    bhh = dir ? bhh_r : bhh_f;
  int ycol0 = dir ? HH : 0;

  // W_hh fragments resident in registers: 3 gates x 4 K-tiles.
  v16h wfrag[3][4];
#pragma unroll
  for (int g = 0; g < 3; ++g)
#pragma unroll
    for (int kt = 0; kt < 4; ++kt)
      wfrag[g][kt] =
          load_b_frag(whh + (size_t)(g * HH + wave * 16) * HH + kt * 32, HH);

  float br = bhh[0 * HH + wave * 16 + col];
  float bz = bhh[1 * HH + wave * 16 + col];
  float bn = bhh[2 * HH + wave * 16 + col];

  __shared__ __align__(16) _Float16 hbuf[2][16 * HH];
  for (int i = threadIdx.x; i < 16 * HH; i += 256) hbuf[0][i] = (_Float16)0.0f;
  __syncthreads();

  v8f hprev = {};

  for (int step = 0; step < TT; ++step) {
    int t  = dir ? (TT - 1 - step) : step;
    int rb = step & 1, wb = rb ^ 1;
    size_t mtile = (size_t)t * BTILES + btile;

    // prefetch this step's gate pre-activations (independent of h)
    v8h xf[3];
#pragma unroll
    for (int g = 0; g < 3; ++g)
      xf[g] = *(const v8h*)(xp + (mtile * NTIL + (g * 8 + wave)) * 256 +
                            lane * 8);

    // recurrent GEMM: hg = h @ W_hh^T  (A-frags from LDS)
    v16h a[4];
#pragma unroll
    for (int kt = 0; kt < 4; ++kt)
      a[kt] = load_a_frag(&hbuf[rb][kt * 32], HH);

    v8f acc[3] = {};
#pragma unroll
    for (int g = 0; g < 3; ++g)
#pragma unroll
      for (int kt = 0; kt < 4; ++kt)
        acc[g] = wmma16(a[kt], wfrag[g][kt], acc[g]);

    // gates + state update
    v8f hnew;
#pragma unroll
    for (int i = 0; i < 8; ++i) {
      float r = fast_sigmoid((float)xf[0][i] + acc[0][i] + br);
      float z = fast_sigmoid((float)xf[1][i] + acc[1][i] + bz);
      float n = fast_tanh((float)xf[2][i] + r * (acc[2][i] + bn));
      hnew[i] = (1.0f - z) * n + z * hprev[i];
    }
    hprev = hnew;

    // publish h_new: LDS (for next step's A-frags) + global y
    _Float16* yp = y + (mtile * 16 + mrow) * 256 + ycol0 + wave * 16 + col;
#pragma unroll
    for (int i = 0; i < 8; ++i) {
      _Float16 hv = (_Float16)hnew[i];
      hbuf[wb][(mrow + i) * HH + wave * 16 + col] = hv;
      yp[(size_t)i * 256] = hv;
    }
    __syncthreads();
  }
}

// --------------------------- output projection -----------------------------
__global__ void out_proj_kernel(const _Float16* __restrict__ rec,
                                const float* __restrict__ out_w,
                                const float* __restrict__ out_b,
                                const float* __restrict__ mean,
                                const float* __restrict__ stdv,
                                float* __restrict__ out) {
  int wave = blockIdx.x * (blockDim.x >> 5) + (threadIdx.x >> 5);
  int lane = threadIdx.x & 31;
  size_t row = (size_t)wave;
  v8h v = *(const v8h*)(rec + row * 256 + lane * 8);
  float s = 0.0f;
#pragma unroll
  for (int i = 0; i < 8; ++i) s += (float)v[i] * out_w[lane * 8 + i];
#pragma unroll
  for (int m = 16; m > 0; m >>= 1) s += __shfl_xor(s, m);
  if (lane == 0) out[row] = (s + out_b[0]) * stdv[0] + mean[0];
}

// ---------------------------------------------------------------------------
extern "C" void kernel_launch(void* const* d_in, const int* in_sizes, int n_in,
                              void* d_out, int out_size, void* d_ws,
                              size_t ws_size, hipStream_t stream) {
  (void)in_sizes; (void)n_in; (void)out_size; (void)ws_size;

  const float* input      = (const float*)d_in[0];
  const float* g_style    = (const float*)d_in[1];
  const float* char_style = (const float*)d_in[2];
  const float* w_ih[4] = {(const float*)d_in[3],  (const float*)d_in[7],
                          (const float*)d_in[11], (const float*)d_in[15]};
  const float* w_hh[4] = {(const float*)d_in[4],  (const float*)d_in[8],
                          (const float*)d_in[12], (const float*)d_in[16]};
  const float* b_ih[4] = {(const float*)d_in[5],  (const float*)d_in[9],
                          (const float*)d_in[13], (const float*)d_in[17]};
  const float* b_hh[4] = {(const float*)d_in[6],  (const float*)d_in[10],
                          (const float*)d_in[14], (const float*)d_in[18]};
  const float* out_w = (const float*)d_in[19];
  const float* out_b = (const float*)d_in[20];
  const float* mean  = (const float*)d_in[21];
  const float* stdv  = (const float*)d_in[22];
  float* out = (float*)d_out;

  // ---- workspace carve-out (f16 everywhere; XP reused for both layers,
  //      Y0 buffer reused as REC for layer 1) ----
  auto aln = [](size_t x) { return (x + 255) & ~(size_t)255; };
  char* ws = (char*)d_ws;
  size_t szXH = (size_t)MROWS * IN0P * 2;
  size_t szXP = (size_t)MTILES * NTIL * 256 * 2;
  size_t szY  = (size_t)MROWS * 256 * 2;
  size_t o = 0;
  _Float16* XH  = (_Float16*)(ws + o); o = aln(o + szXH);
  _Float16* XPf = (_Float16*)(ws + o); o = aln(o + szXP);
  _Float16* XPr = (_Float16*)(ws + o); o = aln(o + szXP);
  _Float16* Y0  = (_Float16*)(ws + o); o = aln(o + szY);   // also REC
  _Float16* WIH[4];
  _Float16* WHH[4];
  int kin[4] = {IN0P, IN0P, IN1, IN1};
  int ksrc[4] = {400, 400, IN1, IN1};
  for (int i = 0; i < 4; ++i) { WIH[i] = (_Float16*)(ws + o); o = aln(o + (size_t)G3 * kin[i] * 2); }
  for (int i = 0; i < 4; ++i) { WHH[i] = (_Float16*)(ws + o); o = aln(o + (size_t)G3 * HH * 2); }

  // ---- weight conversion (fp32 -> packed NxK f16, zero-padded K) ----
  for (int i = 0; i < 4; ++i) {
    int tot = G3 * kin[i];
    cvt_w_kernel<<<(tot + 255) / 256, 256, 0, stream>>>(w_ih[i], WIH[i], G3,
                                                        ksrc[i], kin[i]);
    tot = G3 * HH;
    cvt_w_kernel<<<(tot + 255) / 256, 256, 0, stream>>>(w_hh[i], WHH[i], G3,
                                                        HH, HH);
  }

  // ---- feature build ----
  build_x_kernel<<<MROWS / 8, 256, 0, stream>>>(input, g_style, char_style, XH);

  // ---- layer 0: input projections + scan ----
  gemm_xp_kernel<<<MTILES / 8, 256, 0, stream>>>(XH, WIH[0], b_ih[0], XPf,
                                                 IN0P, MTILES);
  gemm_xp_kernel<<<MTILES / 8, 256, 0, stream>>>(XH, WIH[1], b_ih[1], XPr,
                                                 IN0P, MTILES);
  gru_scan_kernel<<<dim3(BTILES, 2), 256, 0, stream>>>(XPf, XPr, WHH[0],
                                                       WHH[1], b_hh[0],
                                                       b_hh[1], Y0);

  // ---- layer 1: input projections (reusing XP buffers) + scan (REC = Y0) --
  gemm_xp_kernel<<<MTILES / 8, 256, 0, stream>>>(Y0, WIH[2], b_ih[2], XPf, IN1,
                                                 MTILES);
  gemm_xp_kernel<<<MTILES / 8, 256, 0, stream>>>(Y0, WIH[3], b_ih[3], XPr, IN1,
                                                 MTILES);
  gru_scan_kernel<<<dim3(BTILES, 2), 256, 0, stream>>>(XPf, XPr, WHH[2],
                                                       WHH[3], b_hh[2],
                                                       b_hh[3], Y0);

  // ---- output projection ----
  out_proj_kernel<<<MROWS / 8, 256, 0, stream>>>(Y0, out_w, out_b, mean, stdv,
                                                 out);
}
